// Experts_85203561218637
// MI455X (gfx1250) — matrix operand order repaired and checked by reading, not compiled, not run on hardware
//
#include <hip/hip_runtime.h>

// ---------------- problem constants ----------------
#define T_TOK 4096   // BATCH*SEQ
#define DM    1024   // d_model
#define DFF   4096   // d_ff
#define NEXP  8

typedef __attribute__((ext_vector_type(16))) __bf16 v16bf;
typedef __attribute__((ext_vector_type(8)))  __bf16 v8bf;
typedef __attribute__((ext_vector_type(8)))  float  v8f;

union V16U { v16bf v; v8bf h[2]; };

__device__ __forceinline__ __bf16 f2bf(float f) {
  unsigned u = __builtin_bit_cast(unsigned, f);
  u += 0x7FFFu + ((u >> 16) & 1u);           // round-to-nearest-even
  unsigned short s = (unsigned short)(u >> 16);
  return __builtin_bit_cast(__bf16, s);
}

__device__ __forceinline__ float gelu_tanh(float x) {
  float x3 = x * x * x;
  float t = tanhf(0.7978845608028654f * (x + 0.044715f * x3));
  return 0.5f * x * (1.0f + t);
}

// ---------------- prep kernels ----------------
__global__ void moe_zero_counts(int* counts) {
  if (threadIdx.x < NEXP) counts[threadIdx.x] = 0;
}

__global__ void moe_build_perm(const int* __restrict__ disp,
                               int* counts, int* __restrict__ perm) {
  int t = blockIdx.x * blockDim.x + threadIdx.x;
  if (t < T_TOK) {
    int e = disp[t];
    int s = atomicAdd(&counts[e], 1);
    perm[e * T_TOK + s] = t;
  }
}

__global__ void moe_cvt_x(const float* __restrict__ x, __bf16* __restrict__ xb) {
  size_t i = ((size_t)blockIdx.x * blockDim.x + threadIdx.x) * 8;
  float4 f0 = *(const float4*)(x + i);
  float4 f1 = *(const float4*)(x + i + 4);
  v8bf r;
  r[0] = f2bf(f0.x); r[1] = f2bf(f0.y); r[2] = f2bf(f0.z); r[3] = f2bf(f0.w);
  r[4] = f2bf(f1.x); r[5] = f2bf(f1.y); r[6] = f2bf(f1.z); r[7] = f2bf(f1.w);
  *(v8bf*)(xb + i) = r;
}

// ---------------- fused expert GEMM (double-buffered) ----------------
// Phase1 (GELU=true):  h[tok] = gelu(xb[tok] @ W1_e + b1_e)    (bf16 out)
// Phase2 (GELU=false): out[tok] = h[tok] @ W2_e + b2_e          (f32 out)
template <int KD, int ND, bool GELU>
__global__ __launch_bounds__(256) void moe_gemm(
    const __bf16* __restrict__ A,     // [T, KD] bf16, row per token
    const float*  __restrict__ W,     // [E, KD, ND] f32
    const float*  __restrict__ bias,  // [E, ND] f32
    const int*    __restrict__ counts,
    const int*    __restrict__ perm,  // [E, T] token ids
    __bf16*       __restrict__ hout,  // phase-1 output
    float*        __restrict__ yout)  // phase-2 output
{
  constexpr int BM = 128, BN = 128, BK = 32;
  constexpr int SA = 40, SB = 40;       // LDS strides (bf16 elems), 80B => 16B aligned
  constexpr int MT = T_TOK / BM, NT = ND / BN;
  constexpr int KT = KD / BK;

  __shared__ __bf16 As[2][BM * SA];     // A tiles, row-major [m][k]
  __shared__ __bf16 Bs[2][BN * SB];     // B tiles, N-major  [n][k]
  __shared__ int    toks[BM];

  int bx = blockIdx.x;
  int nt = bx % NT;
  int mt = (bx / NT) % MT;
  int e  = bx / (NT * MT);

  int cnt = counts[e];
  int m0  = mt * BM;
  if (m0 >= cnt) return;                // uniform early-exit
  int n0 = nt * BN;

  int tid  = threadIdx.x;
  int wid  = tid >> 5, lane = tid & 31;
  int wm   = wid >> 2, wn = wid & 3;    // 2x4 wave grid -> 64x32 per wave
  int l15  = lane & 15;
  bool hiL = lane >= 16;

  if (tid < BM) {
    int idx = m0 + tid;
    toks[tid] = (idx < cnt) ? perm[e * T_TOK + idx] : -1;
  }
  __syncthreads();

  v8f acc[4][2] = {};

  const int arow = tid >> 1, aseg = tid & 1;        // A staging: 2 thr/row, 16 bf16 each
  const int bk   = tid >> 3, bc0  = (tid & 7) * 16; // B staging: 8 thr/row, 16 cols each

  const int atok = toks[arow];
  const __bf16* asrc0 = A + (size_t)(atok < 0 ? 0 : atok) * KD + aseg * 16;
  const float*  wsrc0 = W + ((size_t)e * KD + bk) * ND + n0 + bc0;

  // ---- prologue: stage tile 0 into buffer 0 ----
  {
    v8bf a0 = *(const v8bf*)(asrc0);
    v8bf a1 = *(const v8bf*)(asrc0 + 8);
    float4 f0 = *(const float4*)(wsrc0);
    float4 f1 = *(const float4*)(wsrc0 + 4);
    float4 f2 = *(const float4*)(wsrc0 + 8);
    float4 f3 = *(const float4*)(wsrc0 + 12);
    *(v8bf*)&As[0][arow * SA + aseg * 16]     = a0;
    *(v8bf*)&As[0][arow * SA + aseg * 16 + 8] = a1;
    float fv[16] = {f0.x, f0.y, f0.z, f0.w, f1.x, f1.y, f1.z, f1.w,
                    f2.x, f2.y, f2.z, f2.w, f3.x, f3.y, f3.z, f3.w};
#pragma unroll
    for (int c = 0; c < 16; ++c)
      Bs[0][(bc0 + c) * SB + bk] = f2bf(fv[c]);
  }
  __syncthreads();

#pragma unroll 1
  for (int kt = 0; kt < KT; ++kt) {
    const int cur  = kt & 1;
    const bool more = (kt + 1) < KT;

    // ---- issue next tile's global loads first (latency hidden by WMMA) ----
    v8bf na0, na1;
    float4 nf0, nf1, nf2, nf3;
    if (more) {
      const __bf16* as = asrc0 + (kt + 1) * BK;
      const float*  wsp = wsrc0 + (size_t)(kt + 1) * BK * ND;
      na0 = *(const v8bf*)(as);
      na1 = *(const v8bf*)(as + 8);
      nf0 = *(const float4*)(wsp);
      nf1 = *(const float4*)(wsp + 4);
      nf2 = *(const float4*)(wsp + 8);
      nf3 = *(const float4*)(wsp + 12);
      if (kt + 2 < KT)  // prefetch weight stream 2 tiles ahead
        __builtin_prefetch(wsrc0 + (size_t)(kt + 2) * BK * ND, 0, 3);
    }

    // ---- fragments per ISA layout + 8x WMMA on current buffer ----
    V16U af[4];
    int akb = hiL ? 8 : 0;               // lanes16-31 hold K 8..15 / 24..31
#pragma unroll
    for (int am = 0; am < 4; ++am) {
      int r = wm * 64 + am * 16 + l15;
      af[am].h[0] = *(const v8bf*)&As[cur][r * SA + akb];      // K 0..7  (8..15)
      af[am].h[1] = *(const v8bf*)&As[cur][r * SA + akb + 16]; // K 16..23 (24..31)
    }
    V16U bf[2];
    int bkb = hiL ? 8 : 0;               // lanes16-31 hold K 8..15 / 24..31
#pragma unroll
    for (int bn = 0; bn < 2; ++bn) {
      int c = wn * 32 + bn * 16 + l15;   // column index
      bf[bn].h[0] = *(const v8bf*)&Bs[cur][c * SB + bkb];
      bf[bn].h[1] = *(const v8bf*)&Bs[cur][c * SB + bkb + 16];
    }
#pragma unroll
    for (int am = 0; am < 4; ++am)
#pragma unroll
      for (int bn = 0; bn < 2; ++bn)
        acc[am][bn] = __builtin_amdgcn_wmma_f32_16x16x32_bf16(
            false, af[am].v, false, bf[bn].v, (short)0, acc[am][bn],
            false, false);

    // ---- convert/store next tile, single barrier per step ----
    if (more) {
      int nxt = cur ^ 1;
      *(v8bf*)&As[nxt][arow * SA + aseg * 16]     = na0;
      *(v8bf*)&As[nxt][arow * SA + aseg * 16 + 8] = na1;
      float fv[16] = {nf0.x, nf0.y, nf0.z, nf0.w, nf1.x, nf1.y, nf1.z, nf1.w,
                      nf2.x, nf2.y, nf2.z, nf2.w, nf3.x, nf3.y, nf3.z, nf3.w};
#pragma unroll
      for (int c = 0; c < 16; ++c)
        Bs[nxt][(bc0 + c) * SB + bk] = f2bf(fv[c]);
      __syncthreads();
    }
  }

  // ---- epilogue: bias (+gelu), scatter by token id ----
#pragma unroll
  for (int bn = 0; bn < 2; ++bn) {
    int col  = n0 + wn * 32 + bn * 16 + l15;
    float bv = bias[(size_t)e * ND + col];
#pragma unroll
    for (int am = 0; am < 4; ++am) {
#pragma unroll
      for (int vi = 0; vi < 8; ++vi) {
        int lrow = wm * 64 + am * 16 + vi + (hiL ? 8 : 0); // C/D layout
        int tok  = toks[lrow];
        if (tok >= 0) {
          float v = acc[am][bn][vi] + bv;
          if (GELU) {
            hout[(size_t)tok * ND + col] = f2bf(gelu_tanh(v));
          } else {
            yout[(size_t)tok * ND + col] = v;
          }
        }
      }
    }
  }
}

// ---------------- host launcher ----------------
extern "C" void kernel_launch(void* const* d_in, const int* in_sizes, int n_in,
                              void* d_out, int out_size, void* d_ws, size_t ws_size,
                              hipStream_t stream) {
  (void)in_sizes; (void)n_in; (void)out_size; (void)ws_size;
  const float* x   = (const float*)d_in[0];
  const float* w1  = (const float*)d_in[1];
  const float* b1  = (const float*)d_in[2];
  const float* w2  = (const float*)d_in[3];
  const float* b2  = (const float*)d_in[4];
  const int*  disp = (const int*)d_in[5];
  float* out = (float*)d_out;

  char* ws = (char*)d_ws;
  int*    counts = (int*)ws;                                   // 8 ints
  int*    perm   = (int*)(ws + 256);                           // E*T ints
  __bf16* xb     = (__bf16*)(ws + 256 + NEXP * T_TOK * 4);     // T*DM bf16
  __bf16* hb     = (__bf16*)(ws + 256 + NEXP * T_TOK * 4 +
                             (size_t)T_TOK * DM * 2);          // T*DFF bf16

  moe_zero_counts<<<1, 32, 0, stream>>>(counts);
  moe_build_perm<<<T_TOK / 256, 256, 0, stream>>>(disp, counts, perm);
  moe_cvt_x<<<(T_TOK * DM / 8) / 256, 256, 0, stream>>>(x, xb);

  moe_gemm<DM, DFF, true>
      <<<NEXP * (T_TOK / 128) * (DFF / 128), 256, 0, stream>>>(
          xb, w1, b1, counts, perm, hb, nullptr);
  moe_gemm<DFF, DM, false>
      <<<NEXP * (T_TOK / 128) * (DM / 128), 256, 0, stream>>>(
          hb, w2, b2, counts, perm, nullptr, out);
}
